// DepthNet_40114994544789
// MI455X (gfx1250) — matrix operand order repaired and checked by reading, not compiled, not run on hardware
//
#include <hip/hip_runtime.h>
#include <hip/hip_bf16.h>

typedef __attribute__((ext_vector_type(16))) __bf16 v16bf;
typedef __attribute__((ext_vector_type(8)))  float  v8f;

union FragBF { v16bf v; uint4 q[2]; };

// Problem constants (from setup_inputs)
constexpr int V = 3, C = 32, H = 96, W = 96, D = 128;
constexpr int HW  = H * W;           // 9216
constexpr int DHW = D * HW;          // 1179648
constexpr int TILES_W = W / 16;      // 6
constexpr int NTAPS = 27;

// ---------------------------------------------------------------------------
// Kernel P: proj[v] = (affine_mat_stage[1+v] @ inv0)[:3]  -> 3x4 per view
// ---------------------------------------------------------------------------
__global__ void proj_kernel(const float* __restrict__ A,
                            const float* __restrict__ Ainv,
                            float* __restrict__ proj) {
    int t = threadIdx.x;
    if (t >= V * 12) return;
    int v = t / 12, rc = t % 12, r = rc / 4, c = rc % 4;
    const float* Av = A + (1 + v) * 16;
    const float* I0 = Ainv; // inverse of mats[0]
    float s = 0.f;
#pragma unroll
    for (int k = 0; k < 4; ++k) s += Av[r * 4 + k] * I0[k * 4 + c];
    proj[v * 12 + r * 4 + c] = s;
}

// ---------------------------------------------------------------------------
// Kernel W: pack (w_feat 8x32x27 | w_prob 1x32x27 | zero-pad) into bf16 B-
// fragments.  B is 32x16 (K x N) per tap; k order used here: lanes 0-15 hold K=0..15 of column
// N=lane, lanes 16-31 hold K=16..31 of column N=lane-16.
// packed[tap][lane][j] with K = j + 16*(lane>>4), N = lane&15.
// ---------------------------------------------------------------------------
__global__ void pack_weights_kernel(const float* __restrict__ wf,
                                    const float* __restrict__ wp,
                                    __hip_bfloat16* __restrict__ packed) {
    int t = blockIdx.x;        // tap 0..26
    int lane = threadIdx.x;    // 0..31
    int oc = lane & 15;
    int half = lane >> 4;
#pragma unroll
    for (int j = 0; j < 16; ++j) {
        int k = j + half * 16;             // input channel
        float val = 0.f;
        if (oc < 8)       val = wf[(oc * C + k) * NTAPS + t];
        else if (oc == 8) val = wp[k * NTAPS + t];
        packed[((size_t)t * 32 + lane) * 16 + j] = __float2bfloat16(val);
    }
}

// ---------------------------------------------------------------------------
// Kernel A: fused homography-warp + bilinear sample + variance reduction.
// One thread per voxel (d,h,w); loops channels; writes bf16 variance in
// DHWC layout (channel-innermost: 32 bf16 = 64B per voxel).
// ---------------------------------------------------------------------------
__global__ __launch_bounds__(96) void warp_var_kernel(
    const float* __restrict__ feat,       // (V,1,C,H,W)
    const float* __restrict__ proj,       // (V,3,4)
    const float* __restrict__ depth_vals, // (1,D,H,W)
    __hip_bfloat16* __restrict__ varbf)   // (D,H,W,C) bf16
{
    const int w = threadIdx.x, h = blockIdx.x, d = blockIdx.y;
    const int pix = h * W + w;
    const float dv = depth_vals[(size_t)d * HW + pix];
    const float fw = (float)w, fh = (float)h;

    int   i00[V], i01[V], i10[V], i11[V];
    float w00[V], w01[V], w10[V], w11[V];
    int msum = 1;

#pragma unroll
    for (int v = 0; v < V; ++v) {
        const float* m = proj + v * 12;
        float p0 = (m[0] * fw + m[1] * fh + m[2])  * dv + m[3];
        float p1 = (m[4] * fw + m[5] * fh + m[6])  * dv + m[7];
        float p2 = (m[8] * fw + m[9] * fh + m[10]) * dv + m[11];
        float inv = 1.0f / p2;
        float xs = p0 * inv, ys = p1 * inv;
        // mask: strict interior (gx,gy in (-1,1))
        if (xs > 0.f && xs < (float)(W - 1) && ys > 0.f && ys < (float)(H - 1)) msum++;
        float x0f = floorf(xs), y0f = floorf(ys);
        float fx = xs - x0f, fy = ys - y0f;
        int x0 = (int)x0f, y0 = (int)y0f;
        int x1 = x0 + 1, y1 = y0 + 1;
        float vx0 = (x0 >= 0 && x0 <= W - 1) ? 1.f : 0.f;
        float vx1 = (x1 >= 0 && x1 <= W - 1) ? 1.f : 0.f;
        float vy0 = (y0 >= 0 && y0 <= H - 1) ? 1.f : 0.f;
        float vy1 = (y1 >= 0 && y1 <= H - 1) ? 1.f : 0.f;
        int cx0 = min(max(x0, 0), W - 1), cx1 = min(max(x1, 0), W - 1);
        int cy0 = min(max(y0, 0), H - 1), cy1 = min(max(y1, 0), H - 1);
        w00[v] = (1.f - fx) * (1.f - fy) * vx0 * vy0;
        w01[v] = (1.f - fx) * fy         * vx0 * vy1;
        w10[v] = fx * (1.f - fy)         * vx1 * vy0;
        w11[v] = fx * fy                 * vx1 * vy1;
        i00[v] = cy0 * W + cx0;
        i01[v] = cy1 * W + cx0;
        i10[v] = cy0 * W + cx1;
        i11[v] = cy1 * W + cx1;
    }

    const float cnt = 1.0f / (float)msum;
    __hip_bfloat16* outp = varbf + ((size_t)d * HW + pix) * C;

#pragma unroll 8
    for (int c = 0; c < C; ++c) {
        float s = 0.f, q = 0.f;
#pragma unroll
        for (int v = 0; v < V; ++v) {
            const float* fp = feat + ((size_t)v * C + c) * HW;
            float val = fp[i00[v]] * w00[v] + fp[i01[v]] * w01[v] +
                        fp[i10[v]] * w10[v] + fp[i11[v]] * w11[v];
            s += val; q += val * val;
        }
        float m1 = s * cnt;
        outp[c] = __float2bfloat16(q * cnt - m1 * m1);
    }
}

// ---------------------------------------------------------------------------
// Kernel B: 3x3x3 conv over the variance volume as WMMA GEMM.
// One wave computes TWO depth-adjacent output tiles (acc0 @ td0, acc1 @
// td0+1): the 4 shared depth planes (td0-1..td0+2) are loaded once and feed
// up to two WMMAs each -> 54 v_wmma per wave on 36 A-fragment loads.
// Weights are staged block-wide into LDS with GLOBAL_LOAD_ASYNC_TO_LDS_B128
// (ASYNCcnt path, no VGPR round-trip).
// ---------------------------------------------------------------------------
__global__ __launch_bounds__(256) void conv_wmma_kernel(
    const __hip_bfloat16* __restrict__ varbf,   // (D,H,W,C) bf16
    const __hip_bfloat16* __restrict__ packedW, // 27 x 32 x 16 bf16 B-frags
    float* __restrict__ cost_feat,              // (8,D,H,W)
    float* __restrict__ prob_logits)            // (D,H,W)
{
    __shared__ __align__(16) __hip_bfloat16 wlds[NTAPS * 32 * 16];
    {   // async stage of weight fragments into LDS (1728 x 16B)
        const int n16 = NTAPS * 32 * 16 * 2 / 16;
        for (int i = threadIdx.x; i < n16; i += blockDim.x) {
            unsigned lds_off =
                (unsigned)(uintptr_t)((const char*)wlds + (size_t)i * 16);
            unsigned long long gaddr =
                (unsigned long long)(uintptr_t)((const char*)packedW + (size_t)i * 16);
            asm volatile("global_load_async_to_lds_b128 %0, %1, off"
                         :: "v"(lds_off), "v"(gaddr) : "memory");
        }
        asm volatile("s_wait_asynccnt 0x0" ::: "memory");
    }
    __syncthreads();

    const int lane = threadIdx.x & 31;
    const int tile = (blockIdx.x * blockDim.x + threadIdx.x) >> 5; // global wave id
    const int tw = tile % TILES_W;
    const int th = (tile / TILES_W) % H;
    const int td0 = (tile / (TILES_W * H)) * 2;      // even depth tile
    const int w0 = tw * 16;
    const int M = lane & 15;
    const int half = lane >> 4;

    __builtin_prefetch(varbf + ((size_t)td0 * HW + th * W + w0) * C, 0, 1);

    v8f acc0 = {};   // output depth td0
    v8f acc1 = {};   // output depth td0+1
#pragma unroll
    for (int p = 0; p < 4; ++p) {          // depth plane dd = td0-1+p
        int dd = td0 + p - 1;
        bool dok = (dd >= 0) && (dd < D);
#pragma unroll
        for (int kh = 0; kh < 3; ++kh) {
            int hh = th + kh - 1;
            bool hok = dok && (hh >= 0) && (hh < H);
#pragma unroll
            for (int kw = 0; kw < 3; ++kw) {
                int ww = w0 + M + kw - 1;
                FragBF a;
                if (hok && ww >= 0 && ww < W) {
                    // A frag (16x32 bf16), k order used here: lanes 0-15 hold K{0-15},
                    // lanes 16-31 hold K{16-31} of row M.
                    const char* row =
                        (const char*)(varbf + ((size_t)(dd * HW + hh * W + ww)) * C);
                    a.q[0] = *(const uint4*)(row + half * 32);
                    a.q[1] = *(const uint4*)(row + 16 + half * 32);
                } else {
                    a.q[0] = uint4{0u, 0u, 0u, 0u};
                    a.q[1] = uint4{0u, 0u, 0u, 0u};
                }
                if (p <= 2) {   // tap kd = p for tile td0
                    int t = (p * 3 + kh) * 3 + kw;
                    FragBF b;
                    const uint4* wp = (const uint4*)(wlds + ((size_t)t * 32 + lane) * 16);
                    b.q[0] = wp[0];
                    b.q[1] = wp[1];
                    acc0 = __builtin_amdgcn_wmma_f32_16x16x32_bf16(
                        false, a.v, false, b.v, (short)0, acc0, false, false);
                }
                if (p >= 1) {   // tap kd = p-1 for tile td0+1
                    int t = ((p - 1) * 3 + kh) * 3 + kw;
                    FragBF b;
                    const uint4* wp = (const uint4*)(wlds + ((size_t)t * 32 + lane) * 16);
                    b.q[0] = wp[0];
                    b.q[1] = wp[1];
                    acc1 = __builtin_amdgcn_wmma_f32_16x16x32_bf16(
                        false, a.v, false, b.v, (short)0, acc1, false, false);
                }
            }
        }
    }

    // D layout: lane l holds N = l%16, VGPR r <-> M = r + 8*(l>=16)
    const int oc = lane & 15;
    const int base0 = td0 * HW + th * W + w0;
    const int base1 = base0 + HW;
#pragma unroll
    for (int r = 0; r < 8; ++r) {
        int m = r + half * 8;
        if (oc < 8) {
            cost_feat[(size_t)oc * DHW + base0 + m] = acc0[r];
            cost_feat[(size_t)oc * DHW + base1 + m] = acc1[r];
        } else if (oc == 8) {
            prob_logits[base0 + m] = acc0[r];
            prob_logits[base1 + m] = acc1[r];
        }
    }
}

// ---------------------------------------------------------------------------
// Kernel C: per-(h,w) column softmax over D, expected depth, 4-window conf.
// ---------------------------------------------------------------------------
__global__ __launch_bounds__(128) void softmax_depth_kernel(
    const float* __restrict__ logits,     // (D,H,W)
    const float* __restrict__ depth_vals, // (D,H,W)
    float* __restrict__ depth_out,        // (H,W)
    float* __restrict__ conf_out,         // (H,W)
    float* __restrict__ prob_out)         // (D,H,W)
{
    __shared__ float sp[D];
    __shared__ float red[D];
    const int pix = blockIdx.x;
    const int dd = threadIdx.x;

    float lg = logits[(size_t)dd * HW + pix];
    red[dd] = lg; __syncthreads();
    for (int s = D / 2; s > 0; s >>= 1) {
        if (dd < s) red[dd] = fmaxf(red[dd], red[dd + s]);
        __syncthreads();
    }
    float mx = red[0]; __syncthreads();

    float e = __expf(lg - mx);
    red[dd] = e; __syncthreads();
    for (int s = D / 2; s > 0; s >>= 1) {
        if (dd < s) red[dd] += red[dd + s];
        __syncthreads();
    }
    float p = e / red[0]; __syncthreads();

    sp[dd] = p;
    prob_out[(size_t)dd * HW + pix] = p;

    red[dd] = p * depth_vals[(size_t)dd * HW + pix]; __syncthreads();
    for (int s = D / 2; s > 0; s >>= 1) {
        if (dd < s) red[dd] += red[dd + s];
        __syncthreads();
    }
    float depth = red[0]; __syncthreads();

    red[dd] = p * (float)dd; __syncthreads();
    for (int s = D / 2; s > 0; s >>= 1) {
        if (dd < s) red[dd] += red[dd + s];
        __syncthreads();
    }
    if (dd == 0) {
        int didx = (int)red[0];
        didx = min(max(didx, 0), D - 1);
        float c = 0.f;
#pragma unroll
        for (int k = -1; k <= 2; ++k) {
            int j = didx + k;
            if (j >= 0 && j < D) c += sp[j];
        }
        depth_out[pix] = depth;
        conf_out[pix]  = fminf(fmaxf(c, 0.f), 1.f);
    }
}

// ---------------------------------------------------------------------------
extern "C" void kernel_launch(void* const* d_in, const int* in_sizes, int n_in,
                              void* d_out, int out_size, void* d_ws, size_t ws_size,
                              hipStream_t stream) {
    const float* features   = (const float*)d_in[0]; // (3,1,32,96,96)
    const float* amat       = (const float*)d_in[1]; // (4,4,4)
    const float* ainv       = (const float*)d_in[2]; // (4,4,4)
    const float* depth_vals = (const float*)d_in[3]; // (1,128,96,96)
    // d_in[4] = imgs (unused by outputs)
    const float* w_feat     = (const float*)d_in[5]; // (8,32,3,3,3)
    const float* w_prob     = (const float*)d_in[6]; // (1,32,3,3,3)
    // d_in[7]=num_depth, d_in[8]=pad (compile-time constants here)

    char* ws = (char*)d_ws;
    float*          proj    = (float*)ws;                       // 144 B
    __hip_bfloat16* packedW = (__hip_bfloat16*)(ws + 256);      // 27648 B
    __hip_bfloat16* varbf   = (__hip_bfloat16*)(ws + 32768);    // 75,497,472 B
    float*          logits  = (float*)(ws + 32768 + (size_t)DHW * C * 2); // 4.7 MB

    float* out       = (float*)d_out;
    float* depth_out = out;                       // 9216
    float* conf_out  = out + HW;                  // 9216
    float* cost_feat = out + 2 * HW;              // 8*DHW
    float* prob_out  = out + 2 * HW + 8 * (size_t)DHW;

    proj_kernel<<<1, 64, 0, stream>>>(amat, ainv, proj);
    pack_weights_kernel<<<NTAPS, 32, 0, stream>>>(w_feat, w_prob, packedW);
    warp_var_kernel<<<dim3(H, D), W, 0, stream>>>(features, proj, depth_vals, varbf);
    conv_wmma_kernel<<<((D / 2) * H * TILES_W) / 8, 256, 0, stream>>>(
        varbf, packedW, cost_feat, logits);
    softmax_depth_kernel<<<HW, D, 0, stream>>>(logits, depth_vals,
                                               depth_out, conf_out, prob_out);
}
